// ClassSR_60894046323052
// MI455X (gfx1250) — matrix operand order repaired and checked by reading, not compiled, not run on hardware
//
#include <hip/hip_runtime.h>
#include <hip/hip_bf16.h>

typedef __bf16 bf16;
typedef __attribute__((ext_vector_type(16))) __bf16        v16bf;
typedef __attribute__((ext_vector_type(8)))  float         v8f;
typedef __attribute__((ext_vector_type(4)))  unsigned int  u32x4;

union FragBF16 { v16bf v; u32x4 q[2]; };

#define OUT_BF16 0
#define OUT_SHUF 1
#define OUT_F32  2

// ---------------------------------------------------------------- f32 -> bf16
__global__ void f2bf_kernel(const float* __restrict__ in, bf16* __restrict__ out, int n) {
    int i = blockIdx.x * 256 + threadIdx.x;
    if (i < n) out[i] = (bf16)in[i];
}

// ---------------------------------------------------------------- generic implicit-GEMM conv (bf16 WMMA, f32 acc)
// GEMM: M = Cout, N = OH*OW (per image), K = Cin*KS*KS.
// 256 threads = 8 waves. Wave w: M-block = w % WM, N-pair = w / WM (two 16-col N blocks).
template<int WM, int WN>
__global__ __launch_bounds__(256) void conv_wmma(
    const bf16*  __restrict__ in,    const bf16* __restrict__ wt,
    const float* __restrict__ bias,
    const bf16*  __restrict__ resid, const float* __restrict__ addf,
    float* __restrict__ outf, bf16* __restrict__ outbf,
    int Cin, int Cout, int H, int W, int OH, int OWshift,
    int KS, int STRIDE, int PAD, int out_mode, int do_lrelu,
    const int* __restrict__ blist, const int* __restrict__ bcnt)
{
    constexpr int BM = WM * 16, BN = WN * 32, BK = 32, BKP = 40; // BKP: +16B row pad vs bank conflicts
    __shared__ __attribute__((aligned(16))) bf16 As[BM * BKP];
    __shared__ __attribute__((aligned(16))) bf16 Bs[BN * BKP];

    int b;
    if (blist) {                       // routed-batch indirection (uniform per block)
        if ((int)blockIdx.z >= *bcnt) return;
        b = blist[blockIdx.z];
    } else {
        b = blockIdx.z;
    }

    const int tid  = threadIdx.x;
    const int lane = tid & 31;
    const int wv   = tid >> 5;
    const int mi   = wv % WM;
    const int nj   = wv / WM;
    const int OW   = 1 << OWshift;
    const int NPIX = OH << OWshift;
    const int m0   = blockIdx.y * BM;
    const int p0   = blockIdx.x * BN;
    const int K    = Cin * KS * KS;
    const int HW   = H * W;

    // staging-column is i-invariant: (tid + i*256) & 31 == tid & 31
    const int colK = tid & 31;
    const int rowT = tid >> 5;

    v8f acc0 = {}; v8f acc1 = {};

    for (int k0 = 0; k0 < K; k0 += BK) {
        const int  kk     = k0 + colK;
        const bool kvalid = (kk < K);
        // decode k -> (ci, kh, kw) ONCE per K-step (hoisted out of slot loops)
        int ci, kh, kw;
        if (KS == 1)      { ci = kk;      kh = 0;          kw = 0; }
        else if (KS == 4) { ci = kk >> 4; int r = kk & 15; kh = r >> 2; kw = r & 3; }
        else              { ci = kk / 9;  int r = kk - ci * 9; kh = r / 3; kw = r - kh * 3; }
        const size_t cbase = (size_t)(b * Cin + ci) * HW;

        // ---- stage A (weights [Cout][K], already GEMM-ready) ----
        #pragma unroll
        for (int i = 0; i < (BM * BK) / 256; ++i) {
            int row = rowT + i * 8;
            int m = m0 + row;
            bf16 v = (bf16)0.0f;
            if (m < Cout && kvalid) v = wt[m * K + kk];
            As[row * BKP + colK] = v;
        }
        // ---- stage B (im2col gather; spatial dims are powers of 2) ----
        #pragma unroll
        for (int i = 0; i < (BN * BK) / 256; ++i) {
            int n = rowT + i * 8;
            int p = p0 + n;
            bf16 v = (bf16)0.0f;
            if (p < NPIX && kvalid) {
                int oy = p >> OWshift, ox = p & (OW - 1);
                int iy = oy * STRIDE + kh - PAD;
                int ix = ox * STRIDE + kw - PAD;
                if ((unsigned)iy < (unsigned)H && (unsigned)ix < (unsigned)W)
                    v = in[cbase + iy * W + ix];
            }
            Bs[n * BKP + colK] = v;
        }
        __syncthreads();

        // ---- per-lane fragments, exactly per CDNA5 16-bit WMMA VGPR layouts ----
        FragBF16 fa, fb0, fb1;
        {
            // A 16x32: lanes 0-15 K {0..7,16..23}; lanes 16-31 K {8..15,24..31}
            const bf16* sa = &As[(mi * 16 + (lane & 15)) * BKP + ((lane & 16) ? 8 : 0)];
            fa.q[0] = *(const u32x4*)(sa);
            fa.q[1] = *(const u32x4*)(sa + 16);
            // B 32x16: lanes 0-15 K {0..7,16..23}; lanes 16-31 K {8..15,24..31}
            const bf16* sb = &Bs[(nj * 32 + (lane & 15)) * BKP + ((lane & 16) ? 8 : 0)];
            fb0.q[0] = *(const u32x4*)(sb);
            fb0.q[1] = *(const u32x4*)(sb + 16);
            const bf16* sb1 = sb + 16 * BKP;
            fb1.q[0] = *(const u32x4*)(sb1);
            fb1.q[1] = *(const u32x4*)(sb1 + 16);
        }
        acc0 = __builtin_amdgcn_wmma_f32_16x16x32_bf16(false, fa.v, false, fb0.v, (short)0, acc0, false, false);
        acc1 = __builtin_amdgcn_wmma_f32_16x16x32_bf16(false, fa.v, false, fb1.v, (short)0, acc1, false, false);
        __syncthreads();
    }

    // ---- epilogue: bias + lrelu + residual + (shuffle) store ----
    const int mbase = mi * 16 + ((lane & 16) ? 8 : 0);   // C/D: VGPR r -> M = r (+8 for hi lanes)
    const int nlane = lane & 15;
    #pragma unroll
    for (int j = 0; j < 2; ++j) {
        const v8f acc = j ? acc1 : acc0;
        int p = p0 + nj * 32 + j * 16 + nlane;
        if (p >= NPIX) continue;
        int oy = p >> OWshift, ox = p & (OW - 1);
        #pragma unroll
        for (int r = 0; r < 8; ++r) {
            int oc = m0 + mbase + r;
            if (oc >= Cout) continue;
            float val = acc[r] + bias[oc];
            if (do_lrelu) val = (val >= 0.f) ? val : 0.1f * val;
            size_t oidx = (size_t)(b * Cout + oc) * NPIX + p;
            if (resid) val += (float)resid[oidx];
            if (addf)  val += addf[oidx];
            if (out_mode == OUT_F32)       outf[oidx]  = val;
            else if (out_mode == OUT_BF16) outbf[oidx] = (bf16)val;
            else { // pixel shuffle r=2: oc -> (c', r1, r2)
                int c2 = oc >> 2, r1 = (oc >> 1) & 1, r2 = oc & 1;
                size_t sidx = ((size_t)(b * (Cout >> 2) + c2) * (OH << 1) + ((oy << 1) + r1))
                                * (size_t)(OW << 1) + ((ox << 1) + r2);
                outbf[sidx] = (bf16)val;
            }
        }
    }
}

// ---------------------------------------------------------------- pool + fc + argmax + routing lists
__global__ void pool_fc_route(const float* __restrict__ f4, const float* __restrict__ fw,
                              const float* __restrict__ fb, float* __restrict__ logits_out,
                              int* __restrict__ counts, int* __restrict__ lists)
{
    __shared__ int sflag[32];
    int b = threadIdx.x;
    if (b < 32) {
        float v[32];
        for (int c = 0; c < 32; ++c) {
            float s = 0.f;
            const float* pp = f4 + ((size_t)(b * 32 + c) << 8);
            for (int p = 0; p < 256; ++p) s += pp[p];
            v[c] = s * (1.0f / 256.0f);
        }
        float best = -3.4e38f; int bi = 0;
        for (int j = 0; j < 3; ++j) {
            float l = fb[j];
            for (int c = 0; c < 32; ++c) l += v[c] * fw[j * 32 + c];
            logits_out[b * 3 + j] = l;
            if (l > best) { best = l; bi = j; }   // strict > keeps first max (jnp.argmax)
        }
        sflag[b] = bi;
    }
    __syncthreads();
    if (threadIdx.x == 0) {
        int cnt[3] = {0, 0, 0};
        for (int bb = 0; bb < 32; ++bb) { int f = sflag[bb]; lists[f * 32 + cnt[f]++] = bb; }
        counts[0] = cnt[0]; counts[1] = cnt[1]; counts[2] = cnt[2];
    }
}

// ---------------------------------------------------------------- host
extern "C" void kernel_launch(void* const* d_in, const int* in_sizes, int n_in,
                              void* d_out, int out_size, void* d_ws, size_t ws_size,
                              hipStream_t stream)
{
    (void)n_in; (void)out_size; (void)ws_size;
    const bool ins = (in_sizes[0] == 32 * 3 * 64 * 64);  // insertion-order flatten vs jax sorted-key flatten

    struct NM { int nf, nb, cfw, cfb, debw, debb, up1w, up1b, up2w, up2b, lastw, lastb;
                int w1[3], b1[3], w2[3], b2[3]; };
    NM nets[3];
    nets[0].nf = 64; nets[0].nb = 3;
    nets[1].nf = 52; nets[1].nb = 2;
    nets[2].nf = 36; nets[2].nb = 1;
    int xi, c0w,c0b,c1w,c1b,c2w,c2b,c3w,c3b,c4w,c4b,fwi,fbi;

    if (ins) { // {'x', 'params':{'cls':{c0w,c0b,...},'net1':{cf_w,cf_b,blocks,deb,up1,up2,last},...}} insertion order
        xi = 0;
        c0w=1;c0b=2;c1w=3;c1b=4;c2w=5;c2b=6;c3w=7;c3b=8;c4w=9;c4b=10;fwi=11;fbi=12;
        int base = 13;
        for (int i = 0; i < 3; ++i) {
            NM& n = nets[i];
            n.cfw = base; n.cfb = base + 1;
            int o = base + 2;
            for (int k = 0; k < n.nb; ++k) { n.w1[k]=o; n.b1[k]=o+1; n.w2[k]=o+2; n.b2[k]=o+3; o += 4; }
            n.debw=o; n.debb=o+1; n.up1w=o+2; n.up1b=o+3; n.up2w=o+4; n.up2b=o+5; n.lastw=o+6; n.lastb=o+7;
            base = o + 8;
        }
    } else {  // jax tree_leaves: dict keys sorted at every level; 'params' < 'x'
        c0b=0;c0w=1;c1b=2;c1w=3;c2b=4;c2w=5;c3b=6;c3w=7;c4b=8;c4w=9;fbi=10;fwi=11;
        int base = 12;
        for (int i = 0; i < 3; ++i) {
            NM& n = nets[i];
            int o = base;  // 'blocks' sorts first
            for (int k = 0; k < n.nb; ++k) { n.w1[k]=o; n.b1[k]=o+1; n.w2[k]=o+2; n.b2[k]=o+3; o += 4; }
            n.cfb=o; n.cfw=o+1; n.debb=o+2; n.debw=o+3; n.lastb=o+4; n.lastw=o+5;
            n.up1b=o+6; n.up1w=o+7; n.up2b=o+8; n.up2w=o+9;
            base = o + 10;
        }
        xi = base; // 66
    }
    auto F = [&](int i) { return (const float*)d_in[i]; };

    // ---- workspace carve (deterministic layout each call) ----
    char* wsb = (char*)d_ws;
    size_t off = 0;
    auto carve = [&](size_t bytes) -> char* {
        char* p = wsb + off; off = (off + bytes + 255) & ~(size_t)255; return p;
    };
    bf16*  xbf    = (bf16*) carve((size_t)393216 * 2);
    bf16*  f0     = (bf16*) carve((size_t)32 * 128 * 256 * 2);
    bf16*  f1     = (bf16*) carve((size_t)32 * 128 * 256 * 2);
    float* f4     = (float*)carve((size_t)32 * 32 * 256 * 4);
    int*   counts = (int*)  carve(16);
    int*   lists  = (int*)  carve(3 * 32 * 4);
    bf16*  featA  = (bf16*) carve((size_t)32 * 64 * 64 * 64 * 2);
    bf16*  featB  = (bf16*) carve((size_t)32 * 64 * 64 * 64 * 2);
    bf16*  ftmp   = (bf16*) carve((size_t)32 * 64 * 64 * 64 * 2);
    bf16*  h1     = (bf16*) carve((size_t)32 * 64 * 128 * 128 * 2);
    bf16*  h2     = (bf16*) carve((size_t)32 * 64 * 256 * 256 * 2);

    auto cvt = [&](int idx, size_t n) -> bf16* {
        bf16* dst = (bf16*)carve(n * 2);
        f2bf_kernel<<<(unsigned)((n + 255) / 256), 256, 0, stream>>>(F(idx), dst, (int)n);
        return dst;
    };

    bf16* Wc0 = cvt(c0w, 128 * 3 * 16);
    bf16* Wc1 = cvt(c1w, 128 * 128);
    bf16* Wc2 = cvt(c2w, 128 * 128);
    bf16* Wc3 = cvt(c3w, 128 * 128);
    bf16* Wc4 = cvt(c4w, 32 * 128);
    struct NetW { bf16 *cf, *deb, *up1, *up2, *last, *w1[3], *w2[3]; } nw[3];
    for (int i = 0; i < 3; ++i) {
        int nf = nets[i].nf;
        nw[i].cf = cvt(nets[i].cfw, (size_t)nf * 27);
        for (int k = 0; k < nets[i].nb; ++k) {
            nw[i].w1[k] = cvt(nets[i].w1[k], (size_t)nf * nf * 9);
            nw[i].w2[k] = cvt(nets[i].w2[k], (size_t)nf * nf * 9);
        }
        nw[i].deb  = cvt(nets[i].debw,  (size_t)3 * nf * 9);
        nw[i].up1  = cvt(nets[i].up1w,  (size_t)4 * nf * nf * 9);
        nw[i].up2  = cvt(nets[i].up2w,  (size_t)4 * nf * nf * 9);
        nw[i].last = cvt(nets[i].lastw, (size_t)3 * nf * 9);
    }

    auto conv = [&](const bf16* in, const bf16* wt, const float* bias,
                    const bf16* resid, const float* addf, float* outf, bf16* outbf,
                    int Cin, int Cout, int H, int W, int OH, int OW,
                    int KS, int STR, int PAD, int mode, int lrelu,
                    const int* bl, const int* bc) {
        int ows = 0; while ((1 << ows) != OW) ++ows;
        int npix = OH * OW;
        if (Cout >= 192) {        // FLOP hot spots (up1/up2 Cout=256/208): BM=128 halves im2col VALU per WMMA
            dim3 g((npix + 31) / 32, (Cout + 127) / 128, 32);
            conv_wmma<8, 1><<<g, 256, 0, stream>>>(in, wt, bias, resid, addf, outf, outbf,
                Cin, Cout, H, W, OH, ows, KS, STR, PAD, mode, lrelu, bl, bc);
        } else if (Cout >= 33) {
            dim3 g((npix + 63) / 64, (Cout + 63) / 64, 32);
            conv_wmma<4, 2><<<g, 256, 0, stream>>>(in, wt, bias, resid, addf, outf, outbf,
                Cin, Cout, H, W, OH, ows, KS, STR, PAD, mode, lrelu, bl, bc);
        } else if (Cout >= 17) {
            dim3 g((npix + 127) / 128, (Cout + 31) / 32, 32);
            conv_wmma<2, 4><<<g, 256, 0, stream>>>(in, wt, bias, resid, addf, outf, outbf,
                Cin, Cout, H, W, OH, ows, KS, STR, PAD, mode, lrelu, bl, bc);
        } else {
            dim3 g((npix + 255) / 256, (Cout + 15) / 16, 32);
            conv_wmma<1, 8><<<g, 256, 0, stream>>>(in, wt, bias, resid, addf, outf, outbf,
                Cin, Cout, H, W, OH, ows, KS, STR, PAD, mode, lrelu, bl, bc);
        }
    };

    float* out_deblur = (float*)d_out;
    float* out_sr     = (float*)d_out + 393216;
    float* out_logits = (float*)d_out + 393216 + 6291456;
    const float* xf   = F(xi);

    // input -> bf16
    f2bf_kernel<<<(393216 + 255) / 256, 256, 0, stream>>>(xf, xbf, 393216);

    // ---- classifier ----
    conv(xbf, Wc0, F(c0b), nullptr, nullptr, nullptr, f0, 3,   128, 64, 64, 16, 16, 4, 4, 0, OUT_BF16, 1, nullptr, nullptr);
    conv(f0,  Wc1, F(c1b), nullptr, nullptr, nullptr, f1, 128, 128, 16, 16, 16, 16, 1, 1, 0, OUT_BF16, 1, nullptr, nullptr);
    conv(f1,  Wc2, F(c2b), nullptr, nullptr, nullptr, f0, 128, 128, 16, 16, 16, 16, 1, 1, 0, OUT_BF16, 1, nullptr, nullptr);
    conv(f0,  Wc3, F(c3b), nullptr, nullptr, nullptr, f1, 128, 128, 16, 16, 16, 16, 1, 1, 0, OUT_BF16, 1, nullptr, nullptr);
    conv(f1,  Wc4, F(c4b), nullptr, nullptr, f4, nullptr, 128, 32, 16, 16, 16, 16, 1, 1, 0, OUT_F32, 0, nullptr, nullptr);
    pool_fc_route<<<1, 32, 0, stream>>>(f4, F(fwi), F(fbi), out_logits, counts, lists);

    // ---- SR nets (each sample only through its routed net) ----
    for (int i = 0; i < 3; ++i) {
        int nf = nets[i].nf;
        const int* bl = lists + i * 32;
        const int* bc = counts + i;
        conv(xbf, nw[i].cf, F(nets[i].cfb), nullptr, nullptr, nullptr, featA,
             3, nf, 64, 64, 64, 64, 3, 1, 1, OUT_BF16, 1, bl, bc);
        bf16* cur = featA; bf16* nxt = featB;
        for (int k = 0; k < nets[i].nb; ++k) {
            conv(cur,  nw[i].w1[k], F(nets[i].b1[k]), nullptr, nullptr, nullptr, ftmp,
                 nf, nf, 64, 64, 64, 64, 3, 1, 1, OUT_BF16, 1, bl, bc);
            conv(ftmp, nw[i].w2[k], F(nets[i].b2[k]), cur, nullptr, nullptr, nxt,
                 nf, nf, 64, 64, 64, 64, 3, 1, 1, OUT_BF16, 0, bl, bc);
            bf16* t = cur; cur = nxt; nxt = t;
        }
        conv(cur, nw[i].deb,  F(nets[i].debb),  nullptr, xf, out_deblur, nullptr,
             nf, 3, 64, 64, 64, 64, 3, 1, 1, OUT_F32, 0, bl, bc);
        conv(cur, nw[i].up1,  F(nets[i].up1b),  nullptr, nullptr, nullptr, h1,
             nf, 4 * nf, 64, 64, 64, 64, 3, 1, 1, OUT_SHUF, 1, bl, bc);
        conv(h1,  nw[i].up2,  F(nets[i].up2b),  nullptr, nullptr, nullptr, h2,
             nf, 4 * nf, 128, 128, 128, 128, 3, 1, 1, OUT_SHUF, 1, bl, bc);
        conv(h2,  nw[i].last, F(nets[i].lastb), nullptr, nullptr, out_sr, nullptr,
             nf, 3, 256, 256, 256, 256, 3, 1, 1, OUT_F32, 0, bl, bc);
    }
}
